// ViT_29248727285949
// MI455X (gfx1250) — matrix operand order repaired and checked by reading, not compiled, not run on hardware
//
#include <hip/hip_runtime.h>
#include <hip/hip_bf16.h>
#include <math.h>

// ---------------- constants ----------------
#define DM   768
#define DFF  3072
#define NH   12
#define DH   64
#define LQ   197            // sequence length (196 patches + CLS)
#define NB   32             // batch
#define NLAY 12
#define NC   1000
#define ROWS (NB * LQ)      // 6304
#define NPATCH 196
#define PROWS (NB * NPATCH) // 6272
#define LPAD 224            // padded seq length for attention (7 chunks of 32)

typedef __attribute__((ext_vector_type(16))) __bf16 v16bf;
typedef __attribute__((ext_vector_type(8)))  float  v8f;

static __device__ __forceinline__ int imin(int a, int b) { return a < b ? a : b; }

// ---------------- WMMA helpers (wave32, gfx1250) ----------------
static __device__ __forceinline__ v8f wmma_bf16(v16bf a, v16bf b, v8f c) {
  return __builtin_amdgcn_wmma_f32_16x16x32_bf16(false, a, false, b, (short)0, c, false, false);
}

// A fragment: 16x32 bf16 tile, row-major source with row stride `stride` (elems).
// Layout (ISA 7.12.2): M = lane%16; half0 holds K{0..7,16..23}, half1 K{8..15,24..31}.
static __device__ __forceinline__ v16bf frag_a(const __bf16* s, int stride, int lane) {
  int m = lane & 15, hfv = lane >> 4;
  union { v16bf v; __bf16 e[16]; } u;
#pragma unroll
  for (int i = 0; i < 16; ++i) {
    int p = i >> 1, q = i & 1;
    int k = (p < 4) ? (p * 2 + q + hfv * 8) : (16 + (p - 4) * 2 + q + hfv * 8);
    u.e[i] = s[m * stride + k];
  }
  return u.v;
}

// B fragment: 32x16 bf16 tile from N-major buffer: s[n*stride + k].
// Layout: N = lane%16; element i -> K = half*8 + i (i<8), 16 + half*8 + (i-8) (i>=8) (+kOff).
static __device__ __forceinline__ v16bf frag_b(const __bf16* s, int stride, int lane, int kOff) {
  int n = lane & 15, hfv = lane >> 4;
  union { v16bf v; __bf16 e[16]; } u;
#pragma unroll
  for (int i = 0; i < 16; ++i) u.e[i] = s[n * stride + kOff + ((i < 8) ? (hfv * 8 + i) : (16 + hfv * 8 + (i - 8)))];
  return u.v;
}

// async 16B global -> LDS copy (ASYNCcnt-tracked, DMA path; ISA 08 §4)
static __device__ __forceinline__ void async_cp16(unsigned ldsOff, const __bf16* gaddr) {
  asm volatile("global_load_async_to_lds_b128 %0, %1, off"
               :: "v"(ldsOff), "v"(gaddr) : "memory");
}
static __device__ __forceinline__ void wait_async0() {
  asm volatile("s_wait_asynccnt 0x0" ::: "memory");
}

// ---------------- prep kernels ----------------
// straight fp32 -> bf16 convert (keeps [N][K] row-major == N-major B operand), 4/thread
__global__ void convB_kernel(const float* __restrict__ W, __bf16* __restrict__ out, int n4) {
  int i = blockIdx.x * blockDim.x + threadIdx.x;
  if (i >= n4) return;
  float4 w = ((const float4*)W)[i];
  union { __bf16 e[4]; uint2 u; } o;
  o.e[0] = (__bf16)w.x; o.e[1] = (__bf16)w.y; o.e[2] = (__bf16)w.z; o.e[3] = (__bf16)w.w;
  ((uint2*)out)[i] = o.u;
}

// x[32,3,224,224] -> patches bf16 [32*196, 768], channel-major patch features
__global__ void patchify_kernel(const float* __restrict__ x, __bf16* __restrict__ out) {
  int idx = blockIdx.x * blockDim.x + threadIdx.x;
  if (idx >= PROWS * DM) return;
  int d = idx % DM;
  int p = (idx / DM) % NPATCH;
  int b = idx / (DM * NPATCH);
  int c = d >> 8, r = (d >> 4) & 15, col = d & 15;
  int py = p / 14, px = p % 14;
  out[idx] = (__bf16)x[(((size_t)(b * 3 + c)) * 224 + py * 16 + r) * 224 + px * 16 + col];
}

// h[b,l,d] = (l==0 ? cls[d] : embF[b,l-1,d]) + posenc(l,d)
__global__ void assemble_h_kernel(const float* __restrict__ embF, const float* __restrict__ cls,
                                  float* __restrict__ h) {
  int idx = blockIdx.x * blockDim.x + threadIdx.x;
  if (idx >= ROWS * DM) return;
  int d = idx % DM;
  int l = (idx / DM) % LQ;
  int b = idx / (DM * LQ);
  float e = (l == 0) ? cls[d] : embF[((size_t)(b * NPATCH + l - 1)) * DM + d];
  int i = d & ~1;  // even dims exponent i, odd dims exponent i-1
  float ang = (float)l * powf(10000.0f, -(float)i / (float)DM);
  float pe = (d & 1) ? cosf(ang) : sinf(ang);
  h[idx] = e + pe;
}

// ---------------- LayerNorm: fp32 rows -> bf16 rows ----------------
__global__ __launch_bounds__(256)
void ln_kernel(const float* __restrict__ in, size_t rowStride,
               const float* __restrict__ g, const float* __restrict__ bb,
               __bf16* __restrict__ out) {
  int row = blockIdx.x;
  const float* x = in + (size_t)row * rowStride;
  __shared__ float red[256];
  int tid = threadIdx.x;
  float s = 0.f;
  for (int i = tid; i < DM; i += 256) s += x[i];
  red[tid] = s; __syncthreads();
  for (int off = 128; off > 0; off >>= 1) {
    if (tid < off) red[tid] += red[tid + off];
    __syncthreads();
  }
  float mean = red[0] / (float)DM;
  __syncthreads();
  float vs = 0.f;
  for (int i = tid; i < DM; i += 256) { float d = x[i] - mean; vs += d * d; }
  red[tid] = vs; __syncthreads();
  for (int off = 128; off > 0; off >>= 1) {
    if (tid < off) red[tid] += red[tid + off];
    __syncthreads();
  }
  float inv = rsqrtf(red[0] / (float)DM + 1e-5f);
  for (int i = tid; i < DM; i += 256)
    out[(size_t)row * DM + i] = (__bf16)((x[i] - mean) * inv * g[i] + bb[i]);
}

// ---------------- generic WMMA GEMM ----------------
// C[M,N] = act(A[M,K] * B^T + bias) (+ resid); A bf16 [M][K], Bw bf16 [N][K] (N-major).
// 256 threads = 8 waves (4 row-groups x 2 col-groups); block tile 128x128; wave 32x64.
// Interior blocks: double-buffered async global->LDS pipeline (ASYNCcnt), 1 barrier/chunk.
// act: 0=none, 1=gelu(erf), 2=tanh. outF (fp32) and/or outB (bf16) written.
__global__ __launch_bounds__(256)
void gemm_kernel(const __bf16* __restrict__ A, const __bf16* __restrict__ Bw,
                 const float* __restrict__ bias, const float* resid,
                 float* outF, __bf16* outB,
                 int M, int N, int K, int act) {
  __shared__ __align__(16) __bf16 As[2][128 * 32];   // [m][k]
  __shared__ __align__(16) __bf16 Bs[2][128 * 32];   // [n][k]
  int tid = threadIdx.x;
  int lane = tid & 31, wid = tid >> 5;
  int wr = wid & 3, wc = wid >> 2;
  int bm = blockIdx.y * 128, bn = blockIdx.x * 128;
  int nc = K >> 5;
  v8f acc[2][4] = {};

  bool interior = (bm + 128 <= M) && (bn + 128 <= N);
  if (interior) {
    // ---- async double-buffered pipeline ----
    auto stage_async = [&](int buf, int k0) {
#pragma unroll
      for (int u = tid; u < 512; u += 256) {
        int r = u >> 2, cc = (u & 3) << 3;
        async_cp16((unsigned)(uintptr_t)&As[buf][r * 32 + cc],
                   A + (size_t)(bm + r) * K + k0 + cc);
        async_cp16((unsigned)(uintptr_t)&Bs[buf][r * 32 + cc],
                   Bw + (size_t)(bn + r) * K + k0 + cc);
      }
    };
    stage_async(0, 0);
    for (int ci = 0; ci < nc; ++ci) {
      int cur = ci & 1;
      wait_async0();
      __syncthreads();  // publish buf[cur]; also proves buf[1-cur] fully consumed
      if (ci + 1 < nc) stage_async(1 - cur, (ci + 1) << 5);
      v16bf af[2], bfr[4];
#pragma unroll
      for (int mi = 0; mi < 2; ++mi)
        af[mi] = frag_a(&As[cur][(wr * 32 + mi * 16) * 32], 32, lane);
#pragma unroll
      for (int ni = 0; ni < 4; ++ni)
        bfr[ni] = frag_b(&Bs[cur][(wc * 64 + ni * 16) * 32], 32, lane, 0);
#pragma unroll
      for (int mi = 0; mi < 2; ++mi)
#pragma unroll
        for (int ni = 0; ni < 4; ++ni)
          acc[mi][ni] = wmma_bf16(af[mi], bfr[ni], acc[mi][ni]);
    }
  } else {
    // ---- guarded synchronous path (edge blocks) ----
    const uint4 zero4 = {0u, 0u, 0u, 0u};
    for (int ci = 0; ci < nc; ++ci) {
      int k0 = ci << 5;
#pragma unroll
      for (int u = tid; u < 512; u += 256) {
        int r = u >> 2, cc = (u & 3) << 3;
        int gm = bm + r;
        uint4 av = zero4;
        if (gm < M) av = *(const uint4*)(A + (size_t)gm * K + k0 + cc);
        *(uint4*)(&As[0][r * 32 + cc]) = av;
        int gn = bn + r;
        uint4 bv = zero4;
        if (gn < N) bv = *(const uint4*)(Bw + (size_t)gn * K + k0 + cc);
        *(uint4*)(&Bs[0][r * 32 + cc]) = bv;
      }
      __syncthreads();
      v16bf af[2], bfr[4];
#pragma unroll
      for (int mi = 0; mi < 2; ++mi)
        af[mi] = frag_a(&As[0][(wr * 32 + mi * 16) * 32], 32, lane);
#pragma unroll
      for (int ni = 0; ni < 4; ++ni)
        bfr[ni] = frag_b(&Bs[0][(wc * 64 + ni * 16) * 32], 32, lane, 0);
#pragma unroll
      for (int mi = 0; mi < 2; ++mi)
#pragma unroll
        for (int ni = 0; ni < 4; ++ni)
          acc[mi][ni] = wmma_bf16(af[mi], bfr[ni], acc[mi][ni]);
      __syncthreads();
    }
  }

  int l15 = lane & 15, hfv = lane >> 4;
#pragma unroll
  for (int mi = 0; mi < 2; ++mi)
#pragma unroll
    for (int ni = 0; ni < 4; ++ni)
#pragma unroll
      for (int v = 0; v < 8; ++v) {
        int r = bm + wr * 32 + mi * 16 + v + 8 * hfv;
        int c = bn + wc * 64 + ni * 16 + l15;
        if (r < M && c < N) {
          float xv = acc[mi][ni][v];
          if (bias) xv += bias[c];
          if (act == 1) xv = 0.5f * xv * (1.0f + erff(xv * 0.70710678118654752f));
          else if (act == 2) xv = tanhf(xv);
          if (resid) xv += resid[(size_t)r * N + c];
          if (outF) outF[(size_t)r * N + c] = xv;
          if (outB) outB[(size_t)r * N + c] = (__bf16)xv;
        }
      }
}

// ---------------- attention (per batch*head, flash-style online softmax) ----------
// Non-standard head split (faithful to reference):
//   q[l,d]=qkv[l, d*NH+h], k[l,d]=qkv[l,(64+d)*NH+h], v[l,d]=qkv[l,(128+d)*NH+h]
// o written to ob[(b*197+l)*768 + h*64 + d] in bf16.
__global__ __launch_bounds__(128)
void attn_kernel(const __bf16* __restrict__ qkv, __bf16* __restrict__ ob) {
  __shared__ __align__(16) __bf16 lds_k[LPAD][DH];    // [m][d]   28672 B
  __shared__ __align__(16) __bf16 lds_vT[DH][LPAD];   // [d][m]   28672 B
  __shared__ __align__(16) __bf16 lds_p[4][16][32];   // per-wave prob tile, 4096 B
  const int b = blockIdx.x / NH;
  const int hh = blockIdx.x % NH;
  const int tid = threadIdx.x;
  const int lane = tid & 31, wid = tid >> 5;
  const int l15 = lane & 15, hfv = lane >> 4;

  // stage K and V^T for this (b,h)  (stride-NH gather forced by reference layout)
  for (int idx = tid; idx < LPAD * DH; idx += 128) {
    int m = idx >> 6, d = idx & 63;
    float kv = 0.f, vv = 0.f;
    if (m < LQ) {
      const __bf16* rowp = qkv + ((size_t)(b * LQ + m)) * (3 * DM) + hh;
      kv = (float)rowp[(DH + d) * NH];
      vv = (float)rowp[(2 * DH + d) * NH];
    }
    lds_k[m][d] = (__bf16)kv;
    lds_vT[d][m] = (__bf16)vv;
  }
  __syncthreads();

  for (int t = 0; t < 4; ++t) {
    int m0 = t * 64 + wid * 16;
    if (m0 >= LQ) continue;  // no barriers below: per-wave work only

    // q fragments for this 16-row tile (K=64 -> two 16x32 frags)
    v16bf qa0, qa1;
    {
      int row = b * LQ + imin(m0 + l15, LQ - 1);
      const __bf16* base = qkv + (size_t)row * (3 * DM) + hh;
      union { v16bf v; __bf16 e[16]; } u0, u1;
#pragma unroll
      for (int i = 0; i < 16; ++i) {
        int p = i >> 1, q = i & 1;
        int kk = (p < 4) ? (p * 2 + q + hfv * 8) : (16 + (p - 4) * 2 + q + hfv * 8);
        u0.e[i] = base[kk * NH];
        u1.e[i] = base[(kk + 32) * NH];
      }
      qa0 = u0.v; qa1 = u1.v;
    }

    float mrun[8], lrun[8];
    v8f oacc[4] = {};
#pragma unroll
    for (int v = 0; v < 8; ++v) { mrun[v] = -1e30f; lrun[v] = 0.f; }

    for (int kk = 0; kk < LPAD; kk += 32) {
      // two 16x16 score subtiles over full K=64
      union { v8f v; float e[8]; } su[2];
#pragma unroll
      for (int st = 0; st < 2; ++st) {
        int cb = kk + st * 16;
        v16bf kb0 = frag_b(&lds_k[cb][0], DH, lane, 0);
        v16bf kb1 = frag_b(&lds_k[cb][0], DH, lane, 32);
        v8f c = {};
        c = wmma_bf16(qa0, kb0, c);
        c = wmma_bf16(qa1, kb1, c);
        su[st].v = c;
      }
      int c0 = kk + l15, c1 = kk + 16 + l15;
      // scale, mask, online softmax (row = v + 8*half is lane-invariant per half)
#pragma unroll
      for (int v = 0; v < 8; ++v) {
        float a0 = (c0 < LQ) ? su[0].e[v] * 0.125f : -1e30f;
        float a1 = (c1 < LQ) ? su[1].e[v] * 0.125f : -1e30f;
        su[0].e[v] = a0; su[1].e[v] = a1;
        float tm = fmaxf(a0, a1);
        tm = fmaxf(tm, __shfl_xor(tm, 1, 32));
        tm = fmaxf(tm, __shfl_xor(tm, 2, 32));
        tm = fmaxf(tm, __shfl_xor(tm, 4, 32));
        tm = fmaxf(tm, __shfl_xor(tm, 8, 32));
        float mnew = fmaxf(mrun[v], tm);
        float corr = __expf(mrun[v] - mnew);
        float p0 = __expf(su[0].e[v] - mnew);
        float p1 = __expf(su[1].e[v] - mnew);
        float ps = p0 + p1;
        ps += __shfl_xor(ps, 1, 32);
        ps += __shfl_xor(ps, 2, 32);
        ps += __shfl_xor(ps, 4, 32);
        ps += __shfl_xor(ps, 8, 32);
        lrun[v] = lrun[v] * corr + ps;
        mrun[v] = mnew;
#pragma unroll
        for (int nt = 0; nt < 4; ++nt) oacc[nt][v] *= corr;
        lds_p[wid][v + 8 * hfv][l15] = (__bf16)p0;
        lds_p[wid][v + 8 * hfv][16 + l15] = (__bf16)p1;
      }
      // o += P(16x32) * V(32x64)
      v16bf pa = frag_a(&lds_p[wid][0][0], 32, lane);
#pragma unroll
      for (int nt = 0; nt < 4; ++nt) {
        v16bf vb = frag_b(&lds_vT[nt * 16][0], LPAD, lane, kk);
        oacc[nt] = wmma_bf16(pa, vb, oacc[nt]);
      }
    }
    // normalize + masked store
#pragma unroll
    for (int v = 0; v < 8; ++v) {
      int rr = m0 + v + 8 * hfv;
      if (rr < LQ) {
        float inv = 1.f / lrun[v];
#pragma unroll
        for (int nt = 0; nt < 4; ++nt) {
          int d = nt * 16 + l15;
          ob[((size_t)(b * LQ + rr)) * DM + hh * DH + d] = (__bf16)(oacc[nt][v] * inv);
        }
      }
    }
  }
}

// ---------------- host driver ----------------
extern "C" void kernel_launch(void* const* d_in, const int* in_sizes, int n_in,
                              void* d_out, int out_size, void* d_ws, size_t ws_size,
                              hipStream_t stream) {
  (void)in_sizes; (void)n_in; (void)out_size; (void)ws_size;
  const float* x      = (const float*)d_in[0];
  const float* w_proj = (const float*)d_in[1];
  const float* b_proj = (const float*)d_in[2];
  const float* cls    = (const float*)d_in[3];
  const float* ln1_g  = (const float*)d_in[4];
  const float* ln1_b  = (const float*)d_in[5];
  const float* w_qkv  = (const float*)d_in[6];
  const float* b_qkv  = (const float*)d_in[7];
  const float* w_o    = (const float*)d_in[8];
  const float* b_o    = (const float*)d_in[9];
  const float* ln2_g  = (const float*)d_in[10];
  const float* ln2_b  = (const float*)d_in[11];
  const float* w_fc1  = (const float*)d_in[12];
  const float* b_fc1  = (const float*)d_in[13];
  const float* w_fc2  = (const float*)d_in[14];
  const float* b_fc2  = (const float*)d_in[15];
  const float* lnf_g  = (const float*)d_in[16];
  const float* lnf_b  = (const float*)d_in[17];
  const float* w_h    = (const float*)d_in[18];
  const float* b_h    = (const float*)d_in[19];
  const float* w_c    = (const float*)d_in[20];
  const float* b_c    = (const float*)d_in[21];

  char* p = (char*)d_ws;
  size_t off = 0;
  auto alloc = [&](size_t bytes) -> void* {
    void* r = p + off;
    off += (bytes + 255) & ~(size_t)255;
    return r;
  };
  // weights kept [N][K] row-major in bf16 (straight convert; B^T == W)
  __bf16* wprojB = (__bf16*)alloc((size_t)DM * DM * 2);
  __bf16* wqkvB  = (__bf16*)alloc((size_t)(3 * DM) * DM * 2);
  __bf16* woB    = (__bf16*)alloc((size_t)DM * DM * 2);
  __bf16* wfc1B  = (__bf16*)alloc((size_t)DFF * DM * 2);
  __bf16* wfc2B  = (__bf16*)alloc((size_t)DM * DFF * 2);
  __bf16* whB    = (__bf16*)alloc((size_t)DM * DM * 2);
  __bf16* wcB    = (__bf16*)alloc((size_t)NC * DM * 2);
  __bf16* patches= (__bf16*)alloc((size_t)PROWS * DM * 2);
  float*  embF   = (float*) alloc((size_t)PROWS * DM * 4);
  float*  h      = (float*) alloc((size_t)ROWS * DM * 4);
  __bf16* y      = (__bf16*)alloc((size_t)ROWS * DM * 2);
  __bf16* qkvb   = (__bf16*)alloc((size_t)ROWS * 3 * DM * 2);
  __bf16* obuf   = (__bf16*)alloc((size_t)ROWS * DM * 2);
  __bf16* fbuf   = (__bf16*)alloc((size_t)ROWS * DFF * 2);
  __bf16* repb   = (__bf16*)alloc((size_t)NB * DM * 2);
  __bf16* hidb   = (__bf16*)alloc((size_t)NB * DM * 2);

  auto cgrid = [](int n) { return dim3((unsigned)((n + 255) / 256)); };
  auto cv = [&](const float* W, __bf16* out, int nelem) {
    int n4 = nelem / 4;
    convB_kernel<<<cgrid(n4), 256, 0, stream>>>(W, out, n4);
  };

  // weight conversion (no transpose needed: W is already [N][K])
  cv(w_proj, wprojB, DM * DM);
  cv(w_qkv,  wqkvB,  3 * DM * DM);
  cv(w_o,    woB,    DM * DM);
  cv(w_fc1,  wfc1B,  DFF * DM);
  cv(w_fc2,  wfc2B,  DM * DFF);
  cv(w_h,    whB,    DM * DM);
  cv(w_c,    wcB,    NC * DM);

  // patch embedding
  patchify_kernel<<<cgrid(PROWS * DM), 256, 0, stream>>>(x, patches);
  {
    dim3 g((DM + 127) / 128, (PROWS + 127) / 128);
    gemm_kernel<<<g, 256, 0, stream>>>(patches, wprojB, b_proj, nullptr, embF, nullptr,
                                       PROWS, DM, DM, 0);
  }
  assemble_h_kernel<<<cgrid(ROWS * DM), 256, 0, stream>>>(embF, cls, h);

  // 12 layers with SHARED weights (faithful to reference)
  for (int layer = 0; layer < NLAY; ++layer) {
    ln_kernel<<<ROWS, 256, 0, stream>>>(h, DM, ln1_g, ln1_b, y);
    {
      dim3 g((3 * DM + 127) / 128, (ROWS + 127) / 128);
      gemm_kernel<<<g, 256, 0, stream>>>(y, wqkvB, b_qkv, nullptr, nullptr, qkvb,
                                         ROWS, 3 * DM, DM, 0);
    }
    attn_kernel<<<NB * NH, 128, 0, stream>>>(qkvb, obuf);
    {
      dim3 g((DM + 127) / 128, (ROWS + 127) / 128);
      gemm_kernel<<<g, 256, 0, stream>>>(obuf, woB, b_o, h, h, nullptr,
                                         ROWS, DM, DM, 0);
    }
    ln_kernel<<<ROWS, 256, 0, stream>>>(h, DM, ln2_g, ln2_b, y);
    {
      dim3 g((DFF + 127) / 128, (ROWS + 127) / 128);
      gemm_kernel<<<g, 256, 0, stream>>>(y, wfc1B, b_fc1, nullptr, nullptr, fbuf,
                                         ROWS, DFF, DM, 1);
    }
    {
      dim3 g((DM + 127) / 128, (ROWS + 127) / 128);
      gemm_kernel<<<g, 256, 0, stream>>>(fbuf, wfc2B, b_fc2, h, h, nullptr,
                                         ROWS, DM, DFF, 0);
    }
  }

  // final LN over CLS rows (row stride 197*768), tanh head, classifier
  ln_kernel<<<NB, 256, 0, stream>>>(h, (size_t)LQ * DM, lnf_g, lnf_b, repb);
  {
    dim3 g((DM + 127) / 128, (NB + 127) / 128);
    gemm_kernel<<<g, 256, 0, stream>>>(repb, whB, b_h, nullptr, nullptr, hidb,
                                       NB, DM, DM, 2);
  }
  {
    dim3 g((NC + 127) / 128, (NB + 127) / 128);
    gemm_kernel<<<g, 256, 0, stream>>>(hidb, wcB, b_c, nullptr, (float*)d_out, nullptr,
                                       NB, NC, DM, 0);
  }
}
